// AttentionLayer_38938173505871
// MI455X (gfx1250) — matrix operand order repaired and checked by reading, compiled, not run on hardware
//
#include <hip/hip_runtime.h>

// ---------------------------------------------------------------------------
// AttentionLayer for MI455X (gfx1250, wave32, WMMA bf16 + TDM async tiles).
//   S=2048, B=2, H=4096, NH=32, NG=8, HD=128, QKV_N = H + 2*NG*HD = 6144
// Pipeline:
//   1) cvt_f32_bf16:  x, W_qkv, W_dense -> bf16 in workspace
//   2) gemm_bf16_wmma<bias,bf16out>: qkv = x @ W_qkv^T + b   (bf16 out)
//      - K-slabs DMA'd into LDS by the Tensor Data Mover (double buffered,
//        TENSORcnt-synchronized), hardware pad -> 80B LDS row pitch
//   3) rope_rearrange (Q and K) + v_rearrange (V transposed [B,NG,HD,S])
//   4) attn_wmma: causal flash attention with online softmax (WMMA bf16)
//   5) gemm_bf16_wmma<nobias,f32out>: out = O @ W_dense^T    (fp32 -> d_out)
// Workspace requirement: ~252 MB.
// ---------------------------------------------------------------------------

#define S_LEN   2048
#define BATCH   2
#define HDIM    4096
#define NHEADS  32
#define NGROUPS 8
#define HEADD   128
#define QKV_N   6144
#define TOK     (S_LEN * BATCH) /* 4096 token rows, t = s*B + b */

typedef __attribute__((ext_vector_type(16))) __bf16         v16bf;
typedef __attribute__((ext_vector_type(8)))  float          v8f;
typedef __attribute__((ext_vector_type(8)))  unsigned short us8;

union FragBF { v16bf v; us8 h[2]; };

__device__ __forceinline__ unsigned short f2bf(float f) {
  unsigned int u = __builtin_bit_cast(unsigned int, f);
  u += 0x7FFFu + ((u >> 16) & 1u);       // round-to-nearest-even
  return (unsigned short)(u >> 16);
}
__device__ __forceinline__ float bf2f(unsigned short h) {
  return __builtin_bit_cast(float, (unsigned int)h << 16);
}

__device__ __forceinline__ v8f wmma_bf16(v16bf a, v16bf b, v8f c) {
  // 8 args: (neg_a, A, neg_b, B, c_mod, C, reuse_a, reuse_b)
  return __builtin_amdgcn_wmma_f32_16x16x32_bf16(false, a, false, b,
                                                 (short)0, c, false, false);
}

// ---------------------------------------------------------------------------
// Tensor Data Mover: async 2D tile load Global -> LDS (gfx1250).
// Tile: 32 bf16 (64B) x 128 rows, row stride = K elements.  D# pad fields
// insert 16B after each 64B row -> LDS row pitch 80B (= 40 halves), matching
// the WMMA fragment readers.  Tracked by TENSORcnt.
// ---------------------------------------------------------------------------
#if defined(__gfx1250__) && __has_builtin(__builtin_amdgcn_tensor_load_to_lds)
#define USE_TDM 1
#else
#define USE_TDM 0
#endif

#if USE_TDM
typedef __attribute__((ext_vector_type(4))) unsigned int v4u;
typedef __attribute__((ext_vector_type(8))) int          v8i;
typedef __attribute__((ext_vector_type(4))) int          v4i;

__device__ __forceinline__ void tdm_load_tile_32x128(unsigned lds_byte_off,
                                                     const unsigned short* gsrc,
                                                     unsigned k_stride) {
  const unsigned long long ga = (unsigned long long)(size_t)gsrc;
  // D# group 0: count=1 | lds_addr | global_addr[56:0] | type=2 ("image")
  const v4u g0 = { 1u,
                   lds_byte_off,
                   (unsigned)(ga & 0xffffffffu),
                   (unsigned)((ga >> 32) & 0x1ffffffu) | (2u << 30) };
  // D# group 1
  const unsigned w0 = (1u << 16)    // data_size = 1 -> 2 bytes
                    | (1u << 20)    // pad_enable
                    | (3u << 22)    // pad_interval: pad after 16 DWORDs (64B)
                    | (3u << 25);   // pad_amount:   4 DWORDs (16B)
  const unsigned td0 = k_stride;    // tensor_dim0 (elements in a row)
  const unsigned td1 = 128u;        // tensor_dim1 (rows)
  const v8i g1 = { (int)w0,
                   (int)((td0 & 0xffffu) << 16),                  // dim0[15:0]
                   (int)((td0 >> 16) | ((td1 & 0xffffu) << 16)),  // dim0 hi | dim1 lo
                   (int)((td1 >> 16) | (32u << 16)),              // dim1 hi | tile_dim0=32
                   (int)128u,                                     // tile_dim1=128, tile_dim2=0
                   (int)k_stride,                                 // dim0_stride[31:0]
                   0, 0 };                                        // stride hi, dim1_stride
  const v4i gz = { 0, 0, 0, 0 };
#if __clang_major__ >= 23
  const v8i gz8 = { 0, 0, 0, 0, 0, 0, 0, 0 };
  __builtin_amdgcn_tensor_load_to_lds(g0, g1, gz, gz, gz8, 0);
#else
  __builtin_amdgcn_tensor_load_to_lds(g0, g1, gz, gz, 0);
#endif
}
#endif  // USE_TDM

// ---------------------------------------------------------------------------
// fp32 -> bf16 bulk convert (vectorized x4)
// ---------------------------------------------------------------------------
__global__ __launch_bounds__(256) void cvt_f32_bf16(
    const float* __restrict__ src, unsigned short* __restrict__ dst, int n4) {
  int i = blockIdx.x * 256 + threadIdx.x;
  if (i >= n4) return;
  float4 v = reinterpret_cast<const float4*>(src)[i];
  ushort4 o;
  o.x = f2bf(v.x); o.y = f2bf(v.y); o.z = f2bf(v.z); o.w = f2bf(v.w);
  reinterpret_cast<ushort4*>(dst)[i] = o;
}

// ---------------------------------------------------------------------------
// Tiled WMMA GEMM:  C[M,N] = A[M,K] @ B[N,K]^T  (both operands K-contiguous)
// Block: 256 threads = 8 waves; block tile 128x128, K step 32 through LDS.
// Wave (wm in 0..3, wn in 0..1) owns a 32x64 sub-tile = 2x4 WMMA 16x16 tiles.
// LDS rows padded to 40 halves (80 B) to spread bank access.
// TDM path: wave 0 DMAs next K-slab (double buffered) while all waves compute.
// ---------------------------------------------------------------------------
template <bool ADD_BIAS, bool OUT_BF16>
__global__ __launch_bounds__(256) void gemm_bf16_wmma(
    const unsigned short* __restrict__ A, const unsigned short* __restrict__ Bw,
    const float* __restrict__ bias, float* __restrict__ Cf,
    unsigned short* __restrict__ Cb, int M, int N, int K) {
  __shared__ __align__(16) unsigned short As[2][128 * 40];
  __shared__ __align__(16) unsigned short Bs[2][128 * 40];

  const int tid   = threadIdx.x;
  const int wave  = tid >> 5;
  const int lane  = tid & 31;
  const int lhalf = lane >> 4;   // 0 or 1 (wave32 half select)
  const int lmod  = lane & 15;
  const int m0 = blockIdx.x * 128;
  const int n0 = blockIdx.y * 128;
  const int wm = wave & 3;       // 4 waves along M (32 rows each)
  const int wn = wave >> 2;      // 2 waves along N (64 cols each)

  v8f acc[2][4];
#pragma unroll
  for (int mi = 0; mi < 2; ++mi)
#pragma unroll
    for (int ni = 0; ni < 4; ++ni) acc[mi][ni] = {};

#if USE_TDM
  const unsigned ldsA0 = (unsigned)(size_t)(const void*)&As[0][0];
  const unsigned ldsA1 = (unsigned)(size_t)(const void*)&As[1][0];
  const unsigned ldsB0 = (unsigned)(size_t)(const void*)&Bs[0][0];
  const unsigned ldsB1 = (unsigned)(size_t)(const void*)&Bs[1][0];
  if (wave == 0) {
    tdm_load_tile_32x128(ldsA0, A + (size_t)m0 * K, (unsigned)K);
    tdm_load_tile_32x128(ldsB0, Bw + (size_t)n0 * K, (unsigned)K);
  }
  int cur = 0;
  for (int k0 = 0; k0 < K; k0 += 32) {
    if (wave == 0) __builtin_amdgcn_s_wait_tensorcnt(0);
    __syncthreads();  // DMA'd slab visible to all waves
    if (wave == 0 && (k0 + 32) < K) {
      // Kick the next slab's DMA while everyone computes on this one.
      tdm_load_tile_32x128(cur ? ldsA0 : ldsA1,
                           A + (size_t)m0 * K + k0 + 32, (unsigned)K);
      tdm_load_tile_32x128(cur ? ldsB0 : ldsB1,
                           Bw + (size_t)n0 * K + k0 + 32, (unsigned)K);
    }
    const unsigned short* as = &As[cur][0];
    const unsigned short* bs = &Bs[cur][0];

    FragBF af[2];
#pragma unroll
    for (int mi = 0; mi < 2; ++mi) {
      const int row = wm * 32 + mi * 16 + lmod;
      af[mi].h[0] = *(const us8*)&as[row * 40 + lhalf * 8];
      af[mi].h[1] = *(const us8*)&as[row * 40 + 16 + lhalf * 8];
    }
    FragBF bfg[4];
#pragma unroll
    for (int ni = 0; ni < 4; ++ni) {
      const int row = wn * 64 + ni * 16 + lmod;
      bfg[ni].h[0] = *(const us8*)&bs[row * 40 + lhalf * 8];
      bfg[ni].h[1] = *(const us8*)&bs[row * 40 + 16 + lhalf * 8];
    }
#pragma unroll
    for (int mi = 0; mi < 2; ++mi)
#pragma unroll
      for (int ni = 0; ni < 4; ++ni)
        acc[mi][ni] = wmma_bf16(af[mi].v, bfg[ni].v, acc[mi][ni]);
    __syncthreads();  // all reads of buf[cur] done before it is re-filled
    cur ^= 1;
  }
#else
  // Fallback: per-thread vector loads global -> LDS (single buffered).
  for (int k0 = 0; k0 < K; k0 += 32) {
#pragma unroll
    for (int i = 0; i < 2; ++i) {
      const int chunk = tid + i * 256;
      const int row = chunk >> 2;
      const int cc  = chunk & 3;
      *(us8*)&As[0][row * 40 + cc * 8] =
          *(const us8*)&A[(size_t)(m0 + row) * K + k0 + cc * 8];
      *(us8*)&Bs[0][row * 40 + cc * 8] =
          *(const us8*)&Bw[(size_t)(n0 + row) * K + k0 + cc * 8];
      if (k0 + 32 < K) {
        __builtin_prefetch(&A[(size_t)(m0 + row) * K + k0 + 32 + cc * 8], 0, 1);
        __builtin_prefetch(&Bw[(size_t)(n0 + row) * K + k0 + 32 + cc * 8], 0, 1);
      }
    }
    __syncthreads();
    FragBF af[2];
#pragma unroll
    for (int mi = 0; mi < 2; ++mi) {
      const int row = wm * 32 + mi * 16 + lmod;
      af[mi].h[0] = *(const us8*)&As[0][row * 40 + lhalf * 8];
      af[mi].h[1] = *(const us8*)&As[0][row * 40 + 16 + lhalf * 8];
    }
    FragBF bfg[4];
#pragma unroll
    for (int ni = 0; ni < 4; ++ni) {
      const int row = wn * 64 + ni * 16 + lmod;
      bfg[ni].h[0] = *(const us8*)&Bs[0][row * 40 + lhalf * 8];
      bfg[ni].h[1] = *(const us8*)&Bs[0][row * 40 + 16 + lhalf * 8];
    }
#pragma unroll
    for (int mi = 0; mi < 2; ++mi)
#pragma unroll
      for (int ni = 0; ni < 4; ++ni)
        acc[mi][ni] = wmma_bf16(af[mi].v, bfg[ni].v, acc[mi][ni]);
    __syncthreads();
  }
#endif

  // Epilogue. C layout: VGPR r -> row r + 8*lhalf, col = lmod.
#pragma unroll
  for (int mi = 0; mi < 2; ++mi) {
    const int rbase = m0 + wm * 32 + mi * 16 + 8 * lhalf;
#pragma unroll
    for (int ni = 0; ni < 4; ++ni) {
      const int col = n0 + wn * 64 + ni * 16 + lmod;
      float bv = 0.0f;
      if constexpr (ADD_BIAS) bv = bias[col];
#pragma unroll
      for (int r = 0; r < 8; ++r) {
        const float v = acc[mi][ni][r] + bv;
        if constexpr (OUT_BF16)
          Cb[(size_t)(rbase + r) * N + col] = f2bf(v);
        else
          Cf[(size_t)(rbase + r) * N + col] = v;
      }
    }
  }
}

// ---------------------------------------------------------------------------
// RoPE + rearrange for Q (nheads=32, colOff=0) and K (nheads=8, colOff=4096).
// qkv: [TOK, 6144] bf16 -> out: [B, nheads, S, HD] bf16 with rope applied.
// rope_cache: (S, 64, 2) fp32.  One thread per (token, head, pair).
// ---------------------------------------------------------------------------
__global__ __launch_bounds__(256) void rope_rearrange(
    const unsigned short* __restrict__ qkv, const float* __restrict__ rope,
    unsigned short* __restrict__ out, int nheads, int colOff, int total) {
  int idx = blockIdx.x * 256 + threadIdx.x;
  if (idx >= total) return;
  const int d2  = idx & 63;
  int tmp       = idx >> 6;
  const int hh  = tmp % nheads;
  const int t   = tmp / nheads;
  const int s   = t >> 1;          // B == 2
  const int b   = t & 1;
  const unsigned short* p = qkv + (size_t)t * QKV_N + colOff + hh * HEADD + 2 * d2;
  const float x0 = bf2f(p[0]);
  const float x1 = bf2f(p[1]);
  const float c  = rope[s * 128 + 2 * d2];
  const float sn = rope[s * 128 + 2 * d2 + 1];
  unsigned short* q =
      out + ((size_t)(b * nheads + hh) * S_LEN + s) * HEADD + 2 * d2;
  q[0] = f2bf(x0 * c - x1 * sn);
  q[1] = f2bf(x1 * c + x0 * sn);
}

// V: qkv cols [5120, 6144) -> Vt [B, NG, HD, S] bf16 (transposed for PV WMMA)
__global__ __launch_bounds__(256) void v_rearrange(
    const unsigned short* __restrict__ qkv, unsigned short* __restrict__ vt,
    int total) {
  int idx = blockIdx.x * 256 + threadIdx.x;
  if (idx >= total) return;
  const int d = idx & 127;
  const int g = (idx >> 7) & 7;
  const int t = idx >> 10;
  const int s = t >> 1;
  const int b = t & 1;
  vt[((size_t)(b * NGROUPS + g) * HEADD + d) * S_LEN + s] =
      qkv[(size_t)t * QKV_N + HDIM + NGROUPS * HEADD + g * HEADD + d];
}

// ---------------------------------------------------------------------------
// Causal flash attention, one wave per 16-row q-tile.
// Q:[B,NH,S,HD]  K:[B,NG,S,HD]  Vt:[B,NG,HD,S]  O:[TOK, NH*HD] (all bf16)
// Per 32-key step: 8 WMMA for scores (2 col tiles x 4 K-slabs of HD),
// online softmax via __shfl_xor over the 16-lane N stripe, P staged through
// per-wave LDS (C-layout -> A-layout), 8 WMMA for P @ V.
// ---------------------------------------------------------------------------
__global__ __launch_bounds__(256) void attn_wmma(
    const unsigned short* __restrict__ Q, const unsigned short* __restrict__ Kc,
    const unsigned short* __restrict__ Vt, unsigned short* __restrict__ O) {
  __shared__ __align__(16) unsigned short Pl[8 * 16 * 32];  // 1KB per wave

  const int tid   = threadIdx.x;
  const int wave  = tid >> 5;
  const int lane  = tid & 31;
  const int lhalf = lane >> 4;
  const int lmod  = lane & 15;
  const int qtile = (blockIdx.x * 8 + wave) * 16;
  const int bh = blockIdx.y;
  const int b  = bh >> 5;      // NH == 32
  const int h  = bh & 31;
  const int g  = h >> 2;       // NH/NG == 4

  const unsigned short* Qp =
      Q + ((size_t)(b * NHEADS + h) * S_LEN + qtile) * HEADD;
  const unsigned short* Kp = Kc + (size_t)(b * NGROUPS + g) * S_LEN * HEADD;
  const unsigned short* Vp = Vt + (size_t)(b * NGROUPS + g) * HEADD * S_LEN;
  unsigned short* Pw = &Pl[wave * 512];

  // Q fragments: 4 slabs of K=32 along HD=128
  FragBF qf[4];
#pragma unroll
  for (int kk = 0; kk < 4; ++kk) {
    qf[kk].h[0] = *(const us8*)&Qp[lmod * HEADD + kk * 32 + lhalf * 8];
    qf[kk].h[1] = *(const us8*)&Qp[lmod * HEADD + kk * 32 + 16 + lhalf * 8];
  }

  v8f acc[8];
  float mrow[8], lrow[8];
#pragma unroll
  for (int i = 0; i < 8; ++i) { acc[i] = {}; mrow[i] = -1e30f; lrow[i] = 0.0f; }

  const float scale = 0.08838834764831845f;  // 1/sqrt(128)
  const int kb = (qtile + 47) >> 5;          // causal: keys < qtile+16

  for (int j = 0; j < kb; ++j) {
    const unsigned short* Kj = Kp + (size_t)(j * 32) * HEADD;
    v8f s0 = {}, s1 = {};
#pragma unroll
    for (int kk = 0; kk < 4; ++kk) {
      FragBF b0, b1;
      b0.h[0] = *(const us8*)&Kj[lmod * HEADD + kk * 32 + lhalf * 8];
      b0.h[1] = *(const us8*)&Kj[lmod * HEADD + kk * 32 + 16 + lhalf * 8];
      b1.h[0] = *(const us8*)&Kj[(16 + lmod) * HEADD + kk * 32 + lhalf * 8];
      b1.h[1] = *(const us8*)&Kj[(16 + lmod) * HEADD + kk * 32 + 16 + lhalf * 8];
      s0 = wmma_bf16(qf[kk].v, b0.v, s0);
      s1 = wmma_bf16(qf[kk].v, b1.v, s1);
    }

    const int col0 = j * 32 + lmod;
    const int col1 = col0 + 16;
#pragma unroll
    for (int r = 0; r < 8; ++r) {
      const int rowi = qtile + r + 8 * lhalf;
      float v0 = s0[r] * scale; if (col0 > rowi) v0 = -1e30f;
      float v1 = s1[r] * scale; if (col1 > rowi) v1 = -1e30f;
      float mx = fmaxf(v0, v1);
#pragma unroll
      for (int o = 1; o < 16; o <<= 1) mx = fmaxf(mx, __shfl_xor(mx, o));
      const float newm  = fmaxf(mrow[r], mx);
      const float alpha = __expf(mrow[r] - newm);
      mrow[r] = newm;
      const float p0 = __expf(v0 - newm);
      const float p1 = __expf(v1 - newm);
      float rs = p0 + p1;
#pragma unroll
      for (int o = 1; o < 16; o <<= 1) rs += __shfl_xor(rs, o);
      lrow[r] = lrow[r] * alpha + rs;
#pragma unroll
      for (int nt = 0; nt < 8; ++nt) acc[nt][r] *= alpha;
      // Stage P (C-layout) to LDS row-major [16][32] bf16
      Pw[(r + 8 * lhalf) * 32 + lmod]      = f2bf(p0);
      Pw[(r + 8 * lhalf) * 32 + 16 + lmod] = f2bf(p1);
    }

    // Reload P as A-fragment (16x32)
    FragBF pf;
    pf.h[0] = *(const us8*)&Pw[lmod * 32 + lhalf * 8];
    pf.h[1] = *(const us8*)&Pw[lmod * 32 + 16 + lhalf * 8];

    // P @ V: Vt row (HD index) is contiguous over keys -> two 16B chunks per lane (keys 8*lhalf.., 16+8*lhalf..)
#pragma unroll
    for (int nt = 0; nt < 8; ++nt) {
      const unsigned short* vrow =
          &Vp[(size_t)(nt * 16 + lmod) * S_LEN + j * 32];
      FragBF vb;
      vb.h[0] = *(const us8*)&vrow[lhalf * 8];
      vb.h[1] = *(const us8*)&vrow[16 + lhalf * 8];
      acc[nt] = wmma_bf16(pf.v, vb.v, acc[nt]);
    }
  }

  // Epilogue: divide by row sum, store O[t, h*128 + col] (t = s*B + b)
#pragma unroll
  for (int r = 0; r < 8; ++r) {
    const float inv = 1.0f / lrow[r];
    const int srow  = qtile + r + 8 * lhalf;
    const size_t t  = (size_t)srow * BATCH + b;
#pragma unroll
    for (int nt = 0; nt < 8; ++nt) {
      const int col = h * HEADD + nt * 16 + lmod;
      O[t * (size_t)HDIM + col] = f2bf(acc[nt][r] * inv);
    }
  }
}

// ---------------------------------------------------------------------------
// Launcher.  Inputs: x, rope_cache, W_qkv, b_qkv, W_dense (all fp32).
// Output: fp32 [S, B, H].  Workspace layout (bytes):
//   xb     @ 0          33,554,432   x in bf16            [4096,4096]
//   wqkvb  @ 33554432   50,331,648   W_qkv bf16           [6144,4096]
//   wdb    @ 83886080   33,554,432   W_dense bf16         [4096,4096]
//   qkvb   @ 117440512  50,331,648   qkv bf16             [4096,6144]
//   Qr     @ 167772160  33,554,432   Q roped              [2,32,2048,128]
//   Kr     @ 201326592   8,388,608   K roped              [2,8,2048,128]
//   Vtb    @ 209715200   8,388,608   V transposed         [2,8,128,2048]
//   Ob     @ 218103808  33,554,432   attention out bf16   [4096,4096]
//   total  251,658,240 bytes
// ---------------------------------------------------------------------------
extern "C" void kernel_launch(void* const* d_in, const int* in_sizes, int n_in,
                              void* d_out, int out_size, void* d_ws,
                              size_t ws_size, hipStream_t stream) {
  const float* x     = (const float*)d_in[0];
  const float* rope  = (const float*)d_in[1];
  const float* Wqkv  = (const float*)d_in[2];
  const float* bqkv  = (const float*)d_in[3];
  const float* Wd    = (const float*)d_in[4];
  float* out         = (float*)d_out;

  char* ws = (char*)d_ws;
  unsigned short* xb    = (unsigned short*)(ws + 0);
  unsigned short* wqkvb = (unsigned short*)(ws + 33554432);
  unsigned short* wdb   = (unsigned short*)(ws + 83886080);
  unsigned short* qkvb  = (unsigned short*)(ws + 117440512);
  unsigned short* Qr    = (unsigned short*)(ws + 167772160);
  unsigned short* Kr    = (unsigned short*)(ws + 201326592);
  unsigned short* Vtb   = (unsigned short*)(ws + 209715200);
  unsigned short* Ob    = (unsigned short*)(ws + 218103808);

  // 1) fp32 -> bf16 conversions
  {
    int n4x = TOK * HDIM / 4;                    // 4,194,304
    int n4w = QKV_N * HDIM / 4;                  // 6,291,456
    cvt_f32_bf16<<<(n4x + 255) / 256, 256, 0, stream>>>(x, xb, n4x);
    cvt_f32_bf16<<<(n4w + 255) / 256, 256, 0, stream>>>(Wqkv, wqkvb, n4w);
    cvt_f32_bf16<<<(n4x + 255) / 256, 256, 0, stream>>>(Wd, wdb, n4x);
  }

  // 2) qkv = x @ W_qkv^T + b  (bf16 out)
  gemm_bf16_wmma<true, true><<<dim3(TOK / 128, QKV_N / 128), 256, 0, stream>>>(
      xb, wqkvb, bqkv, nullptr, qkvb, TOK, QKV_N, HDIM);

  // 3) RoPE (Q, K) + V transpose
  {
    int totQ = TOK * NHEADS * 64;                // 8,388,608 pairs
    int totK = TOK * NGROUPS * 64;               // 2,097,152 pairs
    int totV = TOK * NGROUPS * HEADD;            // 4,194,304 elems
    rope_rearrange<<<(totQ + 255) / 256, 256, 0, stream>>>(qkvb, rope, Qr,
                                                           NHEADS, 0, totQ);
    rope_rearrange<<<(totK + 255) / 256, 256, 0, stream>>>(qkvb, rope, Kr,
                                                           NGROUPS, HDIM, totK);
    v_rearrange<<<(totV + 255) / 256, 256, 0, stream>>>(qkvb, Vtb, totV);
  }

  // 4) Causal flash attention -> Ob [TOK, NH*HD]
  attn_wmma<<<dim3(S_LEN / 128, BATCH * NHEADS), 256, 0, stream>>>(Qr, Kr, Vtb,
                                                                   Ob);

  // 5) out = O @ W_dense^T  (fp32 -> d_out)
  gemm_bf16_wmma<false, false><<<dim3(TOK / 128, HDIM / 128), 256, 0, stream>>>(
      Ob, wdb, nullptr, out, nullptr, TOK, HDIM, HDIM);
}
